// GraphAttentionLayer_31172872634840
// MI455X (gfx1250) — matrix operand order repaired and checked by reading, not compiled, not run on hardware
//
#include <hip/hip_runtime.h>
#include <hip/hip_bf16.h>

typedef __attribute__((ext_vector_type(16))) __bf16 v16bf;
typedef __attribute__((ext_vector_type(8)))  float  v8f;

#define ALPHA 0.2f
#define LOG2E 1.44269504088896340736f
#define BS 4
#define NN 2048
#define FIN 128
#define FOUT 32
#define NH 4

union BFV { v16bf v; __bf16 e[16]; };

// ---------------------------------------------------------------------------
// Kernel 0: one-time transpose+convert of w [H,fin,fout] f32 -> wT [H,fout,fin]
// bf16, so kernel-1 B tiles are single 32-byte contiguous loads.
// ---------------------------------------------------------------------------
__global__ void gat_wcvt_kernel(const float* __restrict__ w,
                                __bf16* __restrict__ wT)
{
  const int idx = blockIdx.x * blockDim.x + threadIdx.x;   // H*FOUT*FIN total
  const int h = idx / (FOUT * FIN);
  const int r = idx % (FOUT * FIN);
  const int o = r / FIN;
  const int k = r % FIN;
  wT[idx] = (__bf16)w[(size_t)h * FIN * FOUT + (size_t)k * FOUT + o];
}

// ---------------------------------------------------------------------------
// Kernel 1: per-head projection h = input @ w[h] via bf16 WMMA (f32 acc).
// Emits hT (bf16, [b,h,fout,n]) + e_src/e_dst = log2(e)*(h.a_src / h.a_dst)
// (f32, pre-scaled so kernel 2 can use raw v_exp_f32 in the log2 domain).
// One wave = one 16-row tile of one head. Block = 4 waves.
// ---------------------------------------------------------------------------
__global__ void gat_proj_kernel(const float* __restrict__ input,
                                const __bf16* __restrict__ wT,
                                const float* __restrict__ a_src,
                                const float* __restrict__ a_dst,
                                __bf16* __restrict__ hT,
                                float* __restrict__ e_src,
                                float* __restrict__ e_dst)
{
  const int wave = threadIdx.x >> 5;
  const int lane = threadIdx.x & 31;
  const int head = blockIdx.y;
  const int rowtile = blockIdx.x * 4 + wave;      // 512 tiles over bs*n rows
  const int b  = rowtile >> 7;                    // 128 tiles per batch
  const int i0 = (rowtile & 127) * 16;

  const int m   = lane & 15;                      // A row / B,C col index
  const int hi  = lane >> 4;
  const int kbA = hi * 16;                        // 16-bit A-layout k base
  const int kbB = hi * 16;                        // 16-bit B-layout k base

  const float*  Arow = input + (size_t)(b * NN + i0 + m) * FIN;
  const __bf16* W0   = wT + (size_t)(head * FOUT + m) * FIN;
  const __bf16* W1   = wT + (size_t)(head * FOUT + 16 + m) * FIN;

  v8f acc0 = {}; v8f acc1 = {};
  #pragma unroll
  for (int ks = 0; ks < 4; ++ks) {                // K = 128 in steps of 32
    const int k0 = ks * 32;
    const float4 A0 = *(const float4*)(Arow + k0 + kbA);
    const float4 A1 = *(const float4*)(Arow + k0 + kbA + 4);
    const float4 A2 = *(const float4*)(Arow + k0 + kbA + 8);
    const float4 A3 = *(const float4*)(Arow + k0 + kbA + 12);
    BFV a;
    a.e[0]=(__bf16)A0.x;  a.e[1]=(__bf16)A0.y;  a.e[2]=(__bf16)A0.z;  a.e[3]=(__bf16)A0.w;
    a.e[4]=(__bf16)A1.x;  a.e[5]=(__bf16)A1.y;  a.e[6]=(__bf16)A1.z;  a.e[7]=(__bf16)A1.w;
    a.e[8]=(__bf16)A2.x;  a.e[9]=(__bf16)A2.y;  a.e[10]=(__bf16)A2.z; a.e[11]=(__bf16)A2.w;
    a.e[12]=(__bf16)A3.x; a.e[13]=(__bf16)A3.y; a.e[14]=(__bf16)A3.z; a.e[15]=(__bf16)A3.w;

    const v16bf b0 = *(const v16bf*)(W0 + k0 + kbB);   // 32B contiguous
    const v16bf b1 = *(const v16bf*)(W1 + k0 + kbB);

    acc0 = __builtin_amdgcn_wmma_f32_16x16x32_bf16(false, a.v, false, b0,
                                                   (short)0, acc0, false, false);
    acc1 = __builtin_amdgcn_wmma_f32_16x16x32_bf16(false, a.v, false, b1,
                                                   (short)0, acc1, false, false);
  }

  // C/D layout: VGPR r -> row (r + 8*hi), col m (tile0) / m+16 (tile1)
  const int bh = b * NH + head;
  __bf16* hTo = hT + (size_t)bh * FOUT * NN;
  float es[8], ed[8];
  #pragma unroll
  for (int r = 0; r < 8; ++r) {
    const int row = i0 + r + 8 * hi;
    hTo[(size_t)m * NN + row]        = (__bf16)acc0[r];
    hTo[(size_t)(m + 16) * NN + row] = (__bf16)acc1[r];
    es[r] = acc0[r] * a_src[head * FOUT + m] + acc1[r] * a_src[head * FOUT + 16 + m];
    ed[r] = acc0[r] * a_dst[head * FOUT + m] + acc1[r] * a_dst[head * FOUT + 16 + m];
  }
  #pragma unroll
  for (int s = 8; s >= 1; s >>= 1) {
    #pragma unroll
    for (int r = 0; r < 8; ++r) {
      es[r] += __shfl_xor(es[r], s, 32);
      ed[r] += __shfl_xor(ed[r], s, 32);
    }
  }
  if (m == 0) {
    #pragma unroll
    for (int r = 0; r < 8; ++r) {
      // pre-scale by log2(e): lrelu commutes with positive scaling, so the
      // whole softmax can run in the log2 domain on raw v_exp_f32
      e_src[(size_t)bh * NN + i0 + r + 8 * hi] = es[r] * LOG2E;
      e_dst[(size_t)bh * NN + i0 + r + 8 * hi] = ed[r] * LOG2E;
    }
  }
}

// ---------------------------------------------------------------------------
// Kernel 1b: maxdst[b,h] = max_j e_dst[b,h,j]  (shift constant for softmax)
// ---------------------------------------------------------------------------
__global__ void gat_maxdst_kernel(const float* __restrict__ e_dst,
                                  float* __restrict__ maxdst)
{
  const int bh = blockIdx.x;
  const int lane = threadIdx.x;
  const float* p = e_dst + (size_t)bh * NN;
  float mx = -3.0e38f;
  for (int j = lane; j < NN; j += 32) mx = fmaxf(mx, p[j]);
  #pragma unroll
  for (int s = 16; s >= 1; s >>= 1) mx = fmaxf(mx, __shfl_xor(mx, s, 32));
  if (lane == 0) maxdst[bh] = mx;
}

// ---------------------------------------------------------------------------
// Kernel 2: fused masked-softmax + attention@h, flash style, single adj pass.
// Block = 4 waves = 4 heads over the same 16 query rows (adj reuse in WGP$).
// e_dst staged in LDS (32 KB of 320 KB); adj prefetched one block ahead.
// Scores are in the log2 domain -> one v_exp_f32 per element, no wrapper mul.
// ---------------------------------------------------------------------------
__global__ void gat_attn_kernel(const int* __restrict__ adj,
                                const __bf16* __restrict__ hT,
                                const float* __restrict__ e_src,
                                const float* __restrict__ e_dst,
                                const float* __restrict__ maxdst,
                                const float* __restrict__ bias,
                                float* __restrict__ out)
{
  __shared__ float ldsED[NH][NN];                  // 32 KB

  const int head = threadIdx.x >> 5;
  const int lane = threadIdx.x & 31;
  const int b  = blockIdx.y;
  const int i0 = blockIdx.x * 16;
  const int bh = b * NH + head;

  const int mA  = lane & 15;
  const int hi  = lane >> 4;
  const int kbA = hi * 16;
  const int kbB = hi * 16;

  // stage this head's e_dst row into LDS (each wave fills its own slice)
  {
    const float* edSrc = e_dst + (size_t)bh * NN;
    #pragma unroll
    for (int j = lane * 4; j < NN; j += 128)
      *(float4*)&ldsED[head][j] = *(const float4*)(edSrc + j);
  }
  __syncthreads();

  const int*    adjRow = adj + ((size_t)b * NN + i0 + mA) * NN;
  const __bf16* hT0 = hT + ((size_t)bh * FOUT + mA) * NN;
  const __bf16* hT1 = hT + ((size_t)bh * FOUT + 16 + mA) * NN;

  const float es = e_src[(size_t)bh * NN + i0 + mA];      // log2 domain
  const float em = es + maxdst[bh];
  const float Mrow = fmaxf(em, ALPHA * em);        // safe shift: exponents <= 0

  v8f acc0 = {}; v8f acc1 = {};
  float rs = 0.f;

  #pragma unroll 2
  for (int jb = 0; jb < NN; jb += 32) {
    // prefetch next adj block of this lane's row (streaming HBM read)
    __builtin_prefetch(adjRow + ((jb + 32 < NN) ? (jb + 32 + kbA) : kbA), 0, 0);

    const int4 a0 = *(const int4*)(adjRow + jb + kbA);
    const int4 a1 = *(const int4*)(adjRow + jb + kbA + 4);
    const int4 a2 = *(const int4*)(adjRow + jb + kbA + 8);
    const int4 a3 = *(const int4*)(adjRow + jb + kbA + 12);
    const float4 e0 = *(const float4*)&ldsED[head][jb + kbA];
    const float4 e1 = *(const float4*)&ldsED[head][jb + kbA + 4];
    const float4 e2 = *(const float4*)&ldsED[head][jb + kbA + 8];
    const float4 e3 = *(const float4*)&ldsED[head][jb + kbA + 12];
    const int   am[16] = {a0.x,a0.y,a0.z,a0.w, a1.x,a1.y,a1.z,a1.w,
                          a2.x,a2.y,a2.z,a2.w, a3.x,a3.y,a3.z,a3.w};
    const float ev[16] = {e0.x,e0.y,e0.z,e0.w, e1.x,e1.y,e1.z,e1.w,
                          e2.x,e2.y,e2.z,e2.w, e3.x,e3.y,e3.z,e3.w};
    BFV p;
    #pragma unroll
    for (int v2 = 0; v2 < 16; ++v2) {
      float e = es + ev[v2];
      e = fmaxf(e, ALPHA * e);                     // leaky relu (log2 domain)
      const float pv = (am[v2] > 0) ? __builtin_amdgcn_exp2f(e - Mrow) : 0.0f;
      rs += pv;
      p.e[v2] = (__bf16)pv;                        // native cvt_pk, RNE
    }
    const v16bf bm0 = *(const v16bf*)(hT0 + jb + kbB);
    const v16bf bm1 = *(const v16bf*)(hT1 + jb + kbB);
    acc0 = __builtin_amdgcn_wmma_f32_16x16x32_bf16(false, p.v, false, bm0,
                                                   (short)0, acc0, false, false);
    acc1 = __builtin_amdgcn_wmma_f32_16x16x32_bf16(false, p.v, false, bm1,
                                                   (short)0, acc1, false, false);
  }

  // full row sum: lane L and L^16 cover disjoint k of the same row
  rs += __shfl_xor(rs, 16, 32);

  const float bias0 = bias[head * FOUT + mA];
  const float bias1 = bias[head * FOUT + 16 + mA];
  #pragma unroll
  for (int r = 0; r < 8; ++r) {
    // C-layout row (r + 8*hi): its sum lives in lane (r + 8*hi) of A-layout
    const float rsum = __shfl(rs, r + 8 * hi, 32);
    const float inv  = 1.0f / fmaxf(rsum, 1e-30f);
    const int row = i0 + r + 8 * hi;
    float o0 = acc0[r] * inv + bias0;
    float o1 = acc1[r] * inv + bias1;
    o0 = (o0 > 0.f) ? o0 : (__builtin_amdgcn_exp2f(o0 * LOG2E) - 1.f);  // ELU
    o1 = (o1 > 0.f) ? o1 : (__builtin_amdgcn_exp2f(o1 * LOG2E) - 1.f);
    float* orow = out + ((size_t)b * NN + row) * (NH * FOUT) + head * FOUT;
    orow[mA]      = o0;
    orow[16 + mA] = o1;
  }
}

// ---------------------------------------------------------------------------
extern "C" void kernel_launch(void* const* d_in, const int* in_sizes, int n_in,
                              void* d_out, int out_size, void* d_ws, size_t ws_size,
                              hipStream_t stream) {
  const float* input = (const float*)d_in[0];
  const int*   adj   = (const int*)d_in[1];
  const float* w     = (const float*)d_in[2];
  const float* a_src = (const float*)d_in[3];
  const float* a_dst = (const float*)d_in[4];
  const float* bias  = (const float*)d_in[5];
  float* out = (float*)d_out;

  char* ws = (char*)d_ws;
  __bf16* hT    = (__bf16*)ws;                                      // 2 MB
  float*  e_src = (float*)(ws + (size_t)BS * NH * FOUT * NN * 2);   // 128 KB
  float*  e_dst = e_src + (size_t)BS * NH * NN;                     // 128 KB
  float*  maxdst = e_dst + (size_t)BS * NH * NN;                    // 64 B
  __bf16* wT    = (__bf16*)(maxdst + 16);                           // 32 KB

  gat_wcvt_kernel<<<(NH * FOUT * FIN) / 256, 256, 0, stream>>>(w, wT);
  dim3 g1(128, NH), b1(128);
  gat_proj_kernel<<<g1, b1, 0, stream>>>(input, wT, a_src, a_dst, hT, e_src, e_dst);
  gat_maxdst_kernel<<<BS * NH, 32, 0, stream>>>(e_dst, maxdst);
  dim3 g2(NN / 16, BS), b2(128);
  gat_attn_kernel<<<g2, b2, 0, stream>>>(adj, hT, e_src, e_dst, maxdst, bias, out);
}
